// PointerNet_28389733826830
// MI455X (gfx1250) — matrix operand order repaired and checked by reading, not compiled, not run on hardware
//
#include <hip/hip_runtime.h>

// ---------------------------------------------------------------------------
// PointerNet (encoder 2-layer biLSTM + pointer-attention decoder) for gfx1250.
// bf16 WMMA (v_wmma_f32_16x16x32_bf16) for every GEMM; persistent scan kernels
// keep weight slices LDS-resident across all timesteps (staged with
// global_load_async_to_lds_b128 / ASYNCcnt) and use a grid-wide atomic barrier
// per step (the recurrence is latency-bound, not throughput-bound).
// ---------------------------------------------------------------------------

typedef __bf16 bf16_t;
typedef __attribute__((ext_vector_type(16))) __bf16 v16bf;
typedef __attribute__((ext_vector_type(8)))  __bf16 v8bf;
typedef __attribute__((ext_vector_type(8)))  float   v8f;

#define NEGV (-10000.0f)

__device__ __forceinline__ float sigmoidf_(float x) { return 1.0f / (1.0f + __expf(-x)); }

// --- WMMA wrapper: D = A(16x32 bf16) * B(32x16 bf16) + C(16x16 f32) --------
__device__ __forceinline__ v8f wmma_bf16(v16bf a, v16bf b, v8f c) {
  return __builtin_amdgcn_wmma_f32_16x16x32_bf16(
      /*neg_a=*/false, a, /*neg_b=*/false, b,
      /*c_mod=*/(short)0, c, /*reuse_a=*/false, /*reuse_b=*/false);
}

// --- CDNA5 async global->LDS copy (16 bytes), tracked by ASYNCcnt ----------
// LDS byte address = low 32 bits of the generic pointer (aperture mapping).
__device__ __forceinline__ void async_copy_b128(void* lds_dst, const void* gsrc) {
  unsigned off = (unsigned)(size_t)lds_dst;
  asm volatile("global_load_async_to_lds_b128 %0, %1, off"
               :: "v"(off), "v"(gsrc)
               : "memory");
}
__device__ __forceinline__ void wait_async0() {
  asm volatile("s_wait_asynccnt 0x0" ::: "memory");
}

// A fragment (16x32 bf16, documented wave32 layout):
// lanes 0-15: row=lane,   elems 0..7 = K k0..k0+7,  elems 8..15 = K k0+16..k0+23
// lanes16-31: row=lane-16, same with k0+8 base.
__device__ __forceinline__ v16bf load_a_frag(const bf16_t* base, int pitch, int k0) {
  int lane = threadIdx.x & 31;
  int row  = lane & 15;
  int ko   = k0 + ((lane >> 4) << 3);
  const bf16_t* p = base + (size_t)row * (size_t)pitch + ko;
  v16bf a;
  *((v8bf*)&a)       = *(const v8bf*)(p);
  *(((v8bf*)&a) + 1) = *(const v8bf*)(p + 16);
  return a;
}

// B fragment (32x16 bf16): lane holds one output column n (row of W[N][K]),
// elems 0..7 = K kb..kb+7, elems 8..15 = K kb+16..kb+23, kb = k0 (+8 for lanes 16-31).
__device__ __forceinline__ v16bf load_b_frag(const bf16_t* base, int pitch, int n0, int k0) {
  int lane = threadIdx.x & 31;
  int col  = n0 + (lane & 15);
  int kb   = k0 + ((lane >> 4) << 3);
  const bf16_t* p = base + (size_t)col * (size_t)pitch + kb;
  v16bf b;
  *((v8bf*)&b)       = *(const v8bf*)(p);
  *(((v8bf*)&b) + 1) = *(const v8bf*)(p + 16);
  return b;
}

// --- grid-wide barrier (persistent kernels; counter zeroed per launch) -----
__device__ __forceinline__ void grid_sync(unsigned* cnt, unsigned target) {
  __syncthreads();
  if (threadIdx.x == 0) {
    __threadfence();
    atomicAdd(cnt, 1u);
    while (atomicAdd(cnt, 0u) < target) { __builtin_amdgcn_s_sleep(2); }
    __threadfence();
  }
  __syncthreads();
}

// ---------------------------------------------------------------------------
// utility kernels
// ---------------------------------------------------------------------------
__global__ void zero_kernel(unsigned* __restrict__ p, size_t nwords) {
  size_t stride = (size_t)gridDim.x * blockDim.x;
  for (size_t i = (size_t)blockIdx.x * blockDim.x + threadIdx.x; i < nwords; i += stride)
    p[i] = 0u;
}

// dst[N][K1+K2] (bf16) = [A | B] (f32). K2==0 -> plain convert of A.
__global__ void concat_w_kernel(const float* __restrict__ A, const float* __restrict__ B,
                                bf16_t* __restrict__ dst, int N, int K1, int K2) {
  int K = K1 + K2;
  size_t total = (size_t)N * K;
  size_t stride = (size_t)gridDim.x * blockDim.x;
  for (size_t i = (size_t)blockIdx.x * blockDim.x + threadIdx.x; i < total; i += stride) {
    int n = (int)(i / K);
    int k = (int)(i - (size_t)n * K);
    float v = (k < K1) ? A[(size_t)n * K1 + k] : B[(size_t)n * K2 + (k - K1)];
    dst[i] = (bf16_t)v;
  }
}

// x0[tok][k] = bf16(emb[ids[tok]][k]), E = 512
__global__ void embed_kernel(const int* __restrict__ ids, const float* __restrict__ emb,
                             bf16_t* __restrict__ outb, int ntok) {
  size_t total = (size_t)ntok * 512;
  size_t stride = (size_t)gridDim.x * blockDim.x;
  for (size_t i = (size_t)blockIdx.x * blockDim.x + threadIdx.x; i < total; i += stride) {
    size_t tok = i >> 9;
    int k = (int)(i & 511);
    outb[i] = (bf16_t)emb[(size_t)ids[tok] * 512 + k];
  }
}

__global__ void dec_init_kernel(const bf16_t* __restrict__ h1F, const bf16_t* __restrict__ h1B,
                                const float* __restrict__ c1F, const float* __restrict__ c1B,
                                bf16_t* __restrict__ hd0, float* __restrict__ cd,
                                int* __restrict__ di) {
  int i = blockIdx.x * blockDim.x + threadIdx.x;
  if (i < 64 * 1024) {
    int b = i >> 10, k = i & 1023;
    hd0[i] = (k < 512) ? h1F[b * 512 + k] : h1B[b * 512 + (k - 512)];
    cd[i]  = (k < 512) ? c1F[b * 512 + k] : c1B[b * 512 + (k - 512)];
  }
  if (i < 64) di[i] = 0;
}

// stage 64 weight rows {g*H + j0 + jj} into LDS (row pitch PW) via async DMA
__device__ __forceinline__ void stage_weights(const bf16_t* __restrict__ W, bf16_t* Wl,
                                              int H, int Kcat, int PW, int j0, int tid) {
  for (int r = tid >> 5; r < 64; r += 8) {
    const bf16_t* src = W + (size_t)((r >> 4) * H + j0 + (r & 15)) * Kcat;
    bf16_t* dstp = Wl + (size_t)r * PW;
    for (int k = (tid & 31) * 8; k < Kcat; k += 32 * 8)
      async_copy_b128(dstp + k, src + k);
  }
  wait_async0();
  __syncthreads();
}

// ---------------------------------------------------------------------------
// Persistent bidirectional LSTM scan (one layer, both directions).
// grid = 64 blocks (32 fwd + 32 bwd), 256 threads (8 waves).
// Block owns 16 hidden units -> 4 gate column-tiles; weights LDS-resident.
// gates = [x_t | h_t] @ Wcat^T + b  (Kcat = Ein + 512)
// ---------------------------------------------------------------------------
__global__ void __launch_bounds__(256, 1)
lstm_scan_kernel(const bf16_t* __restrict__ xin, int Ein, int T,
                 const bf16_t* __restrict__ WF, const bf16_t* __restrict__ WB,
                 const float* __restrict__ bF, const float* __restrict__ bB,
                 bf16_t* hF0, bf16_t* hF1, bf16_t* hB0, bf16_t* hB1,
                 float* cF, float* cB,
                 bf16_t* __restrict__ yout,
                 unsigned* counter) {
  const int H = 512;
  const int Kcat = Ein + H;
  const int PW = Kcat + 8;   // LDS row pitch (bank-conflict pad)
  const int PG = 68;         // gate staging pitch
  extern __shared__ char smem[];
  bf16_t* Wl  = (bf16_t*)smem;
  float* Gbuf = (float*)(smem + (size_t)64 * PW * sizeof(bf16_t));

  const int tid = threadIdx.x;
  const int dir = blockIdx.x >> 5;
  const int blk = blockIdx.x & 31;
  const int j0  = blk * 16;

  const bf16_t* W   = dir ? WB : WF;
  const float* bias = dir ? bB : bF;
  bf16_t* h0 = dir ? hB0 : hF0;
  bf16_t* h1 = dir ? hB1 : hF1;
  float*  cst = dir ? cB : cF;

  stage_weights(W, Wl, H, Kcat, PW, j0, tid);

  const int wv   = tid >> 5;
  const int lane = tid & 31;
  const int mt   = wv >> 1;          // M tile 0..3 (batch rows mt*16..)
  const int nt0  = (wv & 1) * 2;     // this wave's N tiles: nt0, nt0+1
  const int rbase = (lane >> 4) * 8;
  const int ccol  = lane & 15;

  for (int t = 0; t < T; ++t) {
    const int tt = dir ? (T - 1 - t) : t;
    const bf16_t* hread = (t & 1) ? h1 : h0;
    bf16_t* hwrite      = (t & 1) ? h0 : h1;

    v8f acc0 = {}; v8f acc1 = {};
    const bf16_t* abase_x = xin + ((size_t)(mt * 16) * T + tt) * Ein;
    const bf16_t* abase_h = hread + (size_t)(mt * 16) * H;
    const int pitch_x = T * Ein;

    // x-segment of K
#pragma unroll 2
    for (int kc = 0; kc < Ein; kc += 32) {
      v16bf a  = load_a_frag(abase_x, pitch_x, kc);
      v16bf b0 = load_b_frag(Wl, PW, nt0 * 16, kc);
      v16bf b1 = load_b_frag(Wl, PW, nt0 * 16 + 16, kc);
      acc0 = wmma_bf16(a, b0, acc0);
      acc1 = wmma_bf16(a, b1, acc1);
    }
    // h-segment of K (H = 512, compile-time trip count)
#pragma unroll 2
    for (int kc = 0; kc < H; kc += 32) {
      v16bf a  = load_a_frag(abase_h, H, kc);
      v16bf b0 = load_b_frag(Wl, PW, nt0 * 16, Ein + kc);
      v16bf b1 = load_b_frag(Wl, PW, nt0 * 16 + 16, Ein + kc);
      acc0 = wmma_bf16(a, b0, acc0);
      acc1 = wmma_bf16(a, b1, acc1);
    }

#pragma unroll
    for (int v = 0; v < 8; ++v) {
      Gbuf[(mt * 16 + rbase + v) * PG + nt0 * 16 + ccol]        = acc0[v];
      Gbuf[(mt * 16 + rbase + v) * PG + (nt0 + 1) * 16 + ccol]  = acc1[v];
    }
    __syncthreads();

    // cell update for all 64 batch rows x 16 hidden units
    for (int idx = tid; idx < 64 * 16; idx += 256) {
      int b = idx >> 4, jj = idx & 15;
      float gi = Gbuf[b * PG +  0 + jj] + bias[0 * H + j0 + jj];
      float gf = Gbuf[b * PG + 16 + jj] + bias[1 * H + j0 + jj];
      float gg = Gbuf[b * PG + 32 + jj] + bias[2 * H + j0 + jj];
      float go = Gbuf[b * PG + 48 + jj] + bias[3 * H + j0 + jj];
      float c  = cst[b * H + j0 + jj];
      c = sigmoidf_(gf) * c + sigmoidf_(gi) * tanhf(gg);
      float h = sigmoidf_(go) * tanhf(c);
      cst[b * H + j0 + jj] = c;
      bf16_t hb = (bf16_t)h;
      hwrite[b * H + j0 + jj] = hb;
      yout[((size_t)b * T + tt) * 1024 + dir * 512 + j0 + jj] = hb;
    }
    grid_sync(counter, 64u * (unsigned)(t + 1));
  }
}

// ---------------------------------------------------------------------------
// proj = enc @ aW^T + ab, M = B*T = 32768, N = K = 1024, bf16 out.
// grid = 512 x 16 blocks of 64x64 tiles, 8 waves x 2 tiles.
// ---------------------------------------------------------------------------
__global__ void __launch_bounds__(256, 2)
proj_gemm_kernel(const bf16_t* __restrict__ enc, const bf16_t* __restrict__ aWb,
                 const float* __restrict__ ab, bf16_t* __restrict__ proj) {
  const int Kc = 1024, N = 1024;
  const int tid = threadIdx.x, wv = tid >> 5, lane = tid & 31;
  const int mb = blockIdx.x >> 4;
  const int nb = blockIdx.x & 15;
  const int mt = wv >> 1, nt0 = (wv & 1) * 2;
  const int rbase = (lane >> 4) * 8, ccol = lane & 15;
  const int m0g = mb * 64 + mt * 16;
  const int n0g = nb * 64;

  v8f acc0 = {}; v8f acc1 = {};
  const bf16_t* abase = enc + (size_t)m0g * Kc;
#pragma unroll 2
  for (int kc = 0; kc < Kc; kc += 32) {
    v16bf a  = load_a_frag(abase, Kc, kc);
    v16bf b0 = load_b_frag(aWb, Kc, n0g + nt0 * 16, kc);
    v16bf b1 = load_b_frag(aWb, Kc, n0g + nt0 * 16 + 16, kc);
    acc0 = wmma_bf16(a, b0, acc0);
    acc1 = wmma_bf16(a, b1, acc1);
  }
#pragma unroll
  for (int v = 0; v < 8; ++v) {
    int row = m0g + rbase + v;
    int c0  = n0g + nt0 * 16 + ccol;
    int c1  = c0 + 16;
    proj[(size_t)row * N + c0] = (bf16_t)(acc0[v] + ab[c0]);
    proj[(size_t)row * N + c1] = (bf16_t)(acc1[v] + ab[c1]);
  }
}

// ---------------------------------------------------------------------------
// Persistent pointer decoder: 128 steps of LSTM(1024) + attention + argmax.
// grid = 64 blocks (block k owns hidden units 16k..16k+15 in GEMM phase and
// batch row k in the attention phase), 256 threads.
// ---------------------------------------------------------------------------
__global__ void __launch_bounds__(256, 1)
decoder_kernel(const bf16_t* __restrict__ Wd, const float* __restrict__ bd,
               const float* __restrict__ emb_dec, const int* __restrict__ input_seqs,
               const bf16_t* __restrict__ proj,
               bf16_t* hd0, bf16_t* hd1, float* cd,
               bf16_t* ebuf, int* di,
               float* __restrict__ out, int T, int L,
               unsigned* counter) {
  const int H = 1024, Ein = 512, Kcat = 1536;
  const int PW = Kcat + 8, PG = 68;
  extern __shared__ char smem[];
  bf16_t* Wl  = (bf16_t*)smem;
  float* Gbuf = (float*)(smem + (size_t)64 * PW * sizeof(bf16_t));
  float* red  = Gbuf;                 // 256 f32 (reused post-GEMM)
  int*   redi = (int*)(Gbuf + 256);   // 256 i32
  float* hs   = Gbuf + 512;           // 1024 f32 (h row)

  const int tid = threadIdx.x;
  const int blk = blockIdx.x;
  const int j0  = blk * 16;

  stage_weights(Wd, Wl, H, Kcat, PW, j0, tid);

  const int wv = tid >> 5, lane = tid & 31;
  const int mt = wv >> 1, nt0 = (wv & 1) * 2;
  const int rbase = (lane >> 4) * 8, ccol = lane & 15;

  unsigned bar = 0;
  for (int l = 0; l < L; ++l) {
    // phase 1: embedding gather (block -> its batch row)
    {
      int tok = di[blk];
      const float* er = emb_dec + (size_t)tok * Ein;
      for (int k = tid; k < Ein; k += 256) ebuf[blk * Ein + k] = (bf16_t)er[k];
    }
    grid_sync(counter, 64u * (++bar));

    // phase 2: gates = [e | h] @ Wd^T + bd ; cell update
    const bf16_t* hread = (l & 1) ? hd1 : hd0;
    bf16_t* hwrite      = (l & 1) ? hd0 : hd1;
    v8f acc0 = {}; v8f acc1 = {};
    const bf16_t* abase_x = ebuf  + (size_t)(mt * 16) * Ein;
    const bf16_t* abase_h = hread + (size_t)(mt * 16) * H;
#pragma unroll 2
    for (int kc = 0; kc < Ein; kc += 32) {
      v16bf a  = load_a_frag(abase_x, Ein, kc);
      v16bf b0 = load_b_frag(Wl, PW, nt0 * 16, kc);
      v16bf b1 = load_b_frag(Wl, PW, nt0 * 16 + 16, kc);
      acc0 = wmma_bf16(a, b0, acc0);
      acc1 = wmma_bf16(a, b1, acc1);
    }
#pragma unroll 2
    for (int kc = 0; kc < H; kc += 32) {
      v16bf a  = load_a_frag(abase_h, H, kc);
      v16bf b0 = load_b_frag(Wl, PW, nt0 * 16, Ein + kc);
      v16bf b1 = load_b_frag(Wl, PW, nt0 * 16 + 16, Ein + kc);
      acc0 = wmma_bf16(a, b0, acc0);
      acc1 = wmma_bf16(a, b1, acc1);
    }
#pragma unroll
    for (int v = 0; v < 8; ++v) {
      Gbuf[(mt * 16 + rbase + v) * PG + nt0 * 16 + ccol]       = acc0[v];
      Gbuf[(mt * 16 + rbase + v) * PG + (nt0 + 1) * 16 + ccol] = acc1[v];
    }
    __syncthreads();
    for (int idx = tid; idx < 64 * 16; idx += 256) {
      int b = idx >> 4, jj = idx & 15;
      float gi = Gbuf[b * PG +  0 + jj] + bd[0 * H + j0 + jj];
      float gf = Gbuf[b * PG + 16 + jj] + bd[1 * H + j0 + jj];
      float gg = Gbuf[b * PG + 32 + jj] + bd[2 * H + j0 + jj];
      float go = Gbuf[b * PG + 48 + jj] + bd[3 * H + j0 + jj];
      float c  = cd[b * H + j0 + jj];
      c = sigmoidf_(gf) * c + sigmoidf_(gi) * tanhf(gg);
      float h = sigmoidf_(go) * tanhf(c);
      cd[b * H + j0 + jj] = c;
      hwrite[b * H + j0 + jj] = (bf16_t)h;
    }
    grid_sync(counter, 64u * (++bar));

    // phase 3: attention + log_softmax + pointer argmax (block -> batch row)
    {
      const int b = blk;
      for (int k = tid; k < H; k += 256) hs[k] = (float)hwrite[b * H + k];
      __syncthreads();
      float e0 = 0.f, e1 = 0.f;
      const bf16_t* prow0 = proj + ((size_t)b * T + tid) * H;
      const bf16_t* prow1 = proj + ((size_t)b * T + tid + 256) * H;
      for (int k8 = 0; k8 < H; k8 += 8) {
        v8bf p0 = *(const v8bf*)(prow0 + k8);
        v8bf p1 = *(const v8bf*)(prow1 + k8);
#pragma unroll
        for (int i = 0; i < 8; ++i) {
          float hv = hs[k8 + i];
          e0 += hv * (float)p0[i];
          e1 += hv * (float)p1[i];
        }
      }
      if (input_seqs[b * T + tid] == 0)       e0 += NEGV;
      if (input_seqs[b * T + tid + 256] == 0) e1 += NEGV;
      float bv; int bi;
      if (e0 >= e1) { bv = e0; bi = tid; } else { bv = e1; bi = tid + 256; }
      __syncthreads();
      red[tid] = bv; redi[tid] = bi;
      __syncthreads();
      for (int s = 128; s > 0; s >>= 1) {
        if (tid < s) {
          float ov = red[tid + s]; int oi = redi[tid + s];
          if (ov > red[tid] || (ov == red[tid] && oi < redi[tid])) { red[tid] = ov; redi[tid] = oi; }
        }
        __syncthreads();
      }
      float m = red[0]; int topi = redi[0];
      __syncthreads();
      red[tid] = __expf(e0 - m) + __expf(e1 - m);
      __syncthreads();
      for (int s = 128; s > 0; s >>= 1) {
        if (tid < s) red[tid] += red[tid + s];
        __syncthreads();
      }
      float lse = m + __logf(red[0]);
      float* orow = out + ((size_t)b * L + l) * T;
      orow[tid]       = e0 - lse;
      orow[tid + 256] = e1 - lse;
      if (tid == 0) di[b] = input_seqs[b * T + topi];
    }
    grid_sync(counter, 64u * (++bar));
  }
}

// ---------------------------------------------------------------------------
// host
// ---------------------------------------------------------------------------
extern "C" void kernel_launch(void* const* d_in, const int* in_sizes, int n_in,
                              void* d_out, int out_size, void* d_ws, size_t ws_size,
                              hipStream_t stream) {
  (void)in_sizes; (void)n_in; (void)out_size; (void)ws_size;
  const int B = 64, T = 512, L = 128, E = 512;

  const int*   input_seqs = (const int*)d_in[0];
  const float* emb_enc = (const float*)d_in[3];
  const float* emb_dec = (const float*)d_in[4];
  const float* eW0f = (const float*)d_in[5],  *eU0f = (const float*)d_in[6],  *eb0f = (const float*)d_in[7];
  const float* eW0b = (const float*)d_in[8],  *eU0b = (const float*)d_in[9],  *eb0b = (const float*)d_in[10];
  const float* eW1f = (const float*)d_in[11], *eU1f = (const float*)d_in[12], *eb1f = (const float*)d_in[13];
  const float* eW1b = (const float*)d_in[14], *eU1b = (const float*)d_in[15], *eb1b = (const float*)d_in[16];
  const float* dW   = (const float*)d_in[17], *dU   = (const float*)d_in[18], *db   = (const float*)d_in[19];
  const float* aW   = (const float*)d_in[20], *ab   = (const float*)d_in[21];
  float* out = (float*)d_out;

  char* ws = (char*)d_ws;
  size_t off = 0;
  auto alloc = [&](size_t bytes) -> char* {
    char* p = ws + off;
    off = (off + bytes + 255) & ~(size_t)255;
    return p;
  };

  bf16_t* x0   = (bf16_t*)alloc((size_t)B * T * E * 2);
  bf16_t* y0   = (bf16_t*)alloc((size_t)B * T * 1024 * 2);
  bf16_t* enc  = (bf16_t*)alloc((size_t)B * T * 1024 * 2);
  bf16_t* proj = (bf16_t*)alloc((size_t)B * T * 1024 * 2);
  bf16_t* W0F  = (bf16_t*)alloc((size_t)2048 * 1024 * 2);
  bf16_t* W0B  = (bf16_t*)alloc((size_t)2048 * 1024 * 2);
  bf16_t* W1F  = (bf16_t*)alloc((size_t)2048 * 1536 * 2);
  bf16_t* W1B  = (bf16_t*)alloc((size_t)2048 * 1536 * 2);
  bf16_t* Wdc  = (bf16_t*)alloc((size_t)4096 * 1536 * 2);
  bf16_t* aWb  = (bf16_t*)alloc((size_t)1024 * 1024 * 2);

  size_t state_begin = off;
  bf16_t* h0F0 = (bf16_t*)alloc((size_t)64 * 512 * 2);
  bf16_t* h0F1 = (bf16_t*)alloc((size_t)64 * 512 * 2);
  bf16_t* h0B0 = (bf16_t*)alloc((size_t)64 * 512 * 2);
  bf16_t* h0B1 = (bf16_t*)alloc((size_t)64 * 512 * 2);
  float*  c0F  = (float*)alloc((size_t)64 * 512 * 4);
  float*  c0B  = (float*)alloc((size_t)64 * 512 * 4);
  bf16_t* h1F0 = (bf16_t*)alloc((size_t)64 * 512 * 2);
  bf16_t* h1F1 = (bf16_t*)alloc((size_t)64 * 512 * 2);
  bf16_t* h1B0 = (bf16_t*)alloc((size_t)64 * 512 * 2);
  bf16_t* h1B1 = (bf16_t*)alloc((size_t)64 * 512 * 2);
  float*  c1F  = (float*)alloc((size_t)64 * 512 * 4);
  float*  c1B  = (float*)alloc((size_t)64 * 512 * 4);
  bf16_t* hd0  = (bf16_t*)alloc((size_t)64 * 1024 * 2);
  bf16_t* hd1  = (bf16_t*)alloc((size_t)64 * 1024 * 2);
  float*  cd   = (float*)alloc((size_t)64 * 1024 * 4);
  bf16_t* ebuf = (bf16_t*)alloc((size_t)64 * 512 * 2);
  int*    di   = (int*)alloc((size_t)64 * 4);
  unsigned* counters = (unsigned*)alloc(256);
  size_t state_bytes = off - state_begin;

  // zero states + barrier counters (each call, keeps graph replay deterministic)
  zero_kernel<<<256, 256, 0, stream>>>((unsigned*)(ws + state_begin), state_bytes / 4);

  // weight conversion / concatenation to bf16
  concat_w_kernel<<<512, 256, 0, stream>>>(eW0f, eU0f, W0F, 2048, 512, 512);
  concat_w_kernel<<<512, 256, 0, stream>>>(eW0b, eU0b, W0B, 2048, 512, 512);
  concat_w_kernel<<<512, 256, 0, stream>>>(eW1f, eU1f, W1F, 2048, 1024, 512);
  concat_w_kernel<<<512, 256, 0, stream>>>(eW1b, eU1b, W1B, 2048, 1024, 512);
  concat_w_kernel<<<512, 256, 0, stream>>>(dW, dU, Wdc, 4096, 512, 1024);
  concat_w_kernel<<<512, 256, 0, stream>>>(aW, nullptr, aWb, 1024, 1024, 0);

  embed_kernel<<<1024, 256, 0, stream>>>(input_seqs, emb_enc, x0, B * T);

  const size_t lds0 = (size_t)64 * (1024 + 8) * 2 + (size_t)64 * 68 * 4;
  const size_t lds1 = (size_t)64 * (1536 + 8) * 2 + (size_t)64 * 68 * 4;
  (void)hipFuncSetAttribute(reinterpret_cast<const void*>(lstm_scan_kernel),
                            hipFuncAttributeMaxDynamicSharedMemorySize, (int)lds1);
  (void)hipFuncSetAttribute(reinterpret_cast<const void*>(decoder_kernel),
                            hipFuncAttributeMaxDynamicSharedMemorySize, (int)lds1);

  // encoder layer 0 (bidirectional, persistent)
  lstm_scan_kernel<<<64, 256, lds0, stream>>>(x0, 512, T, W0F, W0B, eb0f, eb0b,
                                              h0F0, h0F1, h0B0, h0B1, c0F, c0B,
                                              y0, counters + 0);
  // encoder layer 1 (input = concat(fwd,bwd) of layer 0)
  lstm_scan_kernel<<<64, 256, lds1, stream>>>(y0, 1024, T, W1F, W1B, eb1f, eb1b,
                                              h1F0, h1F1, h1B0, h1B1, c1F, c1B,
                                              enc, counters + 16);
  // attention precompute
  proj_gemm_kernel<<<8192, 256, 0, stream>>>(enc, aWb, ab, proj);
  // decoder init: h,c = concat(final fwd, final bwd) of layer1 (T even -> buf0)
  dec_init_kernel<<<(64 * 1024 + 255) / 256, 256, 0, stream>>>(h1F0, h1B0, c1F, c1B, hd0, cd, di);
  // pointer decoder (persistent)
  decoder_kernel<<<64, 256, lds1, stream>>>(Wdc, db, emb_dec, input_seqs, proj,
                                            hd0, hd1, cd, ebuf, di, out, T, L,
                                            counters + 32);
}
